// RotaryMultiHeadAttention_45389214384481
// MI455X (gfx1250) — compile-verified, matrix operand order checked by reading, not run on hardware
//
#include <hip/hip_runtime.h>

// ---------------------------------------------------------------------------
// RotaryMultiHeadAttention forward for MI455X (gfx1250, wave32, WMMA).
// fp32->bf16 cast (+weight transpose) -> double-buffered async-to-LDS WMMA
// GEMM (Q,K,V proj) -> RoPE + head permute -> flash attention (WMMA QK^T and
// P.V, online softmax, async-to-LDS K/V^T staging) -> WMMA output projection.
// ---------------------------------------------------------------------------

typedef __attribute__((ext_vector_type(16))) __bf16 v16bf;
typedef __attribute__((ext_vector_type(8)))  float  v8f;
typedef __attribute__((ext_vector_type(4)))  int    v4i;

#define BDIM 4
#define NSEQ 2048
#define DMODEL 1024
#define NHEAD 16
#define HDIM 64

// ---- CDNA5 async global->LDS copy (ASYNCcnt-tracked), with sync fallback ---
#if __has_builtin(__builtin_amdgcn_global_load_async_to_lds_b128)
#define HAVE_ASYNC_LDS 1
#else
#define HAVE_ASYNC_LDS 0
#endif

typedef __attribute__((address_space(1))) v4i gv4i;   // global int4*
typedef __attribute__((address_space(3))) v4i lv4i;   // LDS int4*

static __device__ __forceinline__ void async_cp16(const __bf16* g, __bf16* l) {
#if HAVE_ASYNC_LDS
    __builtin_amdgcn_global_load_async_to_lds_b128(
        (gv4i*)(unsigned long long)g, (lv4i*)l, 0, 0);
#else
    *(uint4*)l = *(const uint4*)g;
#endif
}

static __device__ __forceinline__ void async_wait0() {
#if HAVE_ASYNC_LDS
#if __has_builtin(__builtin_amdgcn_s_wait_asynccnt)
    __builtin_amdgcn_s_wait_asynccnt(0);
#else
    asm volatile("s_wait_asynccnt 0x0" ::: "memory");
#endif
#endif
}

static __device__ __forceinline__ v8f v8f_zero() {
    v8f z = {0.f, 0.f, 0.f, 0.f, 0.f, 0.f, 0.f, 0.f};
    return z;
}

// ---------------------------------------------------------------------------
// fp32 -> bf16 elementwise cast
// ---------------------------------------------------------------------------
__global__ void k_to_bf16(const float* __restrict__ src,
                          __bf16* __restrict__ dst, int n) {
    int i = blockIdx.x * blockDim.x + threadIdx.x;
    if (i < n) dst[i] = (__bf16)src[i];
}

// ---------------------------------------------------------------------------
// fp32 [K][N] -> bf16 transposed [N][K], LDS-tiled 32x32. Grid (32,32),
// block (32,8). Makes the GEMM B operand contiguous along K so tiles can be
// staged with straight async-to-LDS copies (no in-LDS transpose).
// ---------------------------------------------------------------------------
__global__ __launch_bounds__(256)
void k_transpose_bf16(const float* __restrict__ src, __bf16* __restrict__ dst) {
    __shared__ __bf16 t[32][33];
    const int x0 = blockIdx.x * 32, y0 = blockIdx.y * 32;
#pragma unroll
    for (int yy = threadIdx.y; yy < 32; yy += 8)
        t[yy][threadIdx.x] =
            (__bf16)src[(size_t)(y0 + yy) * DMODEL + x0 + threadIdx.x];
    __syncthreads();
#pragma unroll
    for (int yy = threadIdx.y; yy < 32; yy += 8)
        dst[(size_t)(x0 + yy) * DMODEL + y0 + threadIdx.x] = t[threadIdx.x][yy];
}

// ---------------------------------------------------------------------------
// WMMA GEMM: C(M x N) = A(M x K) * Bt(N x K)^T + bias.
// A, Bt row-major bf16 (both contiguous along K). Block 256 threads (8 waves),
// tile 128x128, BK=32; waves 4x2, each wave 32x64 = 2x4 WMMA tiles.
// Double-buffered LDS, tiles staged with async global->LDS b128.
// ---------------------------------------------------------------------------
#define GBM 128
#define GBN 128
#define GBK 32

__global__ __launch_bounds__(256)
void k_gemm_bf16_wmma(const __bf16* __restrict__ A,
                      const __bf16* __restrict__ Bt,
                      const float* __restrict__ bias,
                      void* __restrict__ Cout,
                      int M, int Nn, int K, int outBf16) {
    __shared__ __align__(16) __bf16 sA[2][GBM][GBK];   // 2 x 8 KiB
    __shared__ __align__(16) __bf16 sB[2][GBN][GBK];   // 2 x 8 KiB

    const int tid  = threadIdx.x;
    const int wave = tid >> 5;
    const int lane = tid & 31;
    const int half = lane >> 4;
    const int l16  = lane & 15;
    const int wm   = wave >> 1;          // 0..3
    const int wn   = wave & 1;           // 0..1
    const int bm0  = blockIdx.y * GBM;
    const int bn0  = blockIdx.x * GBN;

    v8f acc[2][4];
#pragma unroll
    for (int i = 0; i < 2; ++i)
#pragma unroll
        for (int j = 0; j < 4; ++j) acc[i][j] = v8f_zero();

    // Stage one 128x32 A tile + 128x32 B tile: 1024 x 16B chunks, 4/thread.
    auto stage = [&](int buf, int k0) {
#pragma unroll
        for (int c = tid; c < 512; c += 256) {
            const int r = c >> 2, cg = (c & 3) * 8;
            async_cp16(A + (size_t)(bm0 + r) * K + k0 + cg, &sA[buf][r][cg]);
        }
#pragma unroll
        for (int c = tid; c < 512; c += 256) {
            const int r = c >> 2, cg = (c & 3) * 8;
            async_cp16(Bt + (size_t)(bn0 + r) * K + k0 + cg, &sB[buf][r][cg]);
        }
    };

    stage(0, 0);
    async_wait0();
    __syncthreads();

    int cur = 0;
    for (int k0 = 0; k0 < K; k0 += GBK) {
        if (k0 + GBK < K) stage(cur ^ 1, k0 + GBK);

        v16bf afr[2], bfr[4];
#pragma unroll
        for (int i = 0; i < 2; ++i)
            afr[i] = *(const v16bf*)&sA[cur][wm * 32 + i * 16 + l16][half * 16];
#pragma unroll
        for (int j = 0; j < 4; ++j)
            bfr[j] = *(const v16bf*)&sB[cur][wn * 64 + j * 16 + l16][half * 16];
#pragma unroll
        for (int i = 0; i < 2; ++i)
#pragma unroll
            for (int j = 0; j < 4; ++j)
                acc[i][j] = __builtin_amdgcn_wmma_f32_16x16x32_bf16(
                    false, afr[i], false, bfr[j], (short)0, acc[i][j],
                    false, false);

        async_wait0();
        __syncthreads();
        cur ^= 1;
    }

    // C layout: VGPR r -> (M = r + 8*half, N = l16) within each 16x16 tile.
#pragma unroll
    for (int i = 0; i < 2; ++i) {
#pragma unroll
        for (int j = 0; j < 4; ++j) {
            const int col = bn0 + wn * 64 + j * 16 + l16;
            const float bb = bias[col];
#pragma unroll
            for (int r = 0; r < 8; ++r) {
                const int row = bm0 + wm * 32 + i * 16 + r + half * 8;
                const float v = acc[i][j][r] + bb;
                if (outBf16)
                    ((__bf16*)Cout)[(size_t)row * Nn + col] = (__bf16)v;
                else
                    ((float*)Cout)[(size_t)row * Nn + col] = v;
            }
        }
    }
}

// ---------------------------------------------------------------------------
// RoPE + permute (B,N,H,HD) -> (B,H,N,HD) for Q and K (bf16 in/out).
// ---------------------------------------------------------------------------
__global__ __launch_bounds__(64)
void k_rope_permute(const __bf16* __restrict__ Qin,
                    const __bf16* __restrict__ Kin,
                    __bf16* __restrict__ Qout,
                    __bf16* __restrict__ Kout) {
    const int t   = threadIdx.x;      // 0..63 head-dim index
    const int idx = blockIdx.x;       // over B*H*N
    const int n   = idx % NSEQ;
    const int bh  = idx / NSEQ;
    const int h   = bh % NHEAD;
    const int b   = bh / NHEAD;

    const int j = t & 31;
    const float inv = __powf(10000.0f, -(float)j * (1.0f / 32.0f));
    float s, c;
    __sincosf((float)n * inv, &s, &c);

    const size_t src = ((size_t)b * NSEQ + n) * DMODEL + h * HDIM;
    const float sgn = (t < 32) ? -1.0f : 1.0f;

    const float q  = (float)Qin[src + t];
    const float qp = (float)Qin[src + (t ^ 32)];
    const float kk = (float)Kin[src + t];
    const float kp = (float)Kin[src + (t ^ 32)];

    const size_t dst = ((size_t)bh * NSEQ + n) * HDIM + t;
    Qout[dst] = (__bf16)(q * c + sgn * qp * s);
    Kout[dst] = (__bf16)(kk * c + sgn * kp * s);
}

// ---------------------------------------------------------------------------
// V permute (B,N,D) -> head-transposed (B,H,HD,N), bf16, so flash attention
// can stage V^T tiles with straight async copies.
// ---------------------------------------------------------------------------
__global__ void k_permute_vt(const __bf16* __restrict__ Vin,
                             __bf16* __restrict__ Vt, int total) {
    int i = blockIdx.x * blockDim.x + threadIdx.x;
    if (i >= total) return;
    const int d  = i % DMODEL;
    const int bn = i / DMODEL;
    const int n  = bn % NSEQ;
    const int b  = bn / NSEQ;
    const int h  = d / HDIM;
    const int hd = d % HDIM;
    Vt[(((size_t)b * NHEAD + h) * HDIM + hd) * NSEQ + n] = Vin[i];
}

// ---------------------------------------------------------------------------
// Flash attention per (b,h). Grid (N/64, B*H), block 128 (4 waves), each wave
// owns 16 query rows. K blocks of 32 double-buffered in LDS via async copies
// (K row-major [32][64], V pre-transposed [64][32]). S = Q.K^T (2 WMMA
// k-steps), online softmax (fp32, 16-lane xor reductions), P re-fragmented
// through per-wave LDS tile, O += P.V (1 WMMA k-step x 4 col tiles).
// ---------------------------------------------------------------------------
__global__ __launch_bounds__(128)
void k_flash_attn_wmma(const __bf16* __restrict__ Q,
                       const __bf16* __restrict__ K,
                       const __bf16* __restrict__ Vt,
                       __bf16* __restrict__ O) {
    __shared__ __align__(16) __bf16 sK[2][32][HDIM];   // 2 x 4 KiB
    __shared__ __align__(16) __bf16 sVt[2][HDIM][32];  // 2 x 4 KiB
    __shared__ __align__(16) __bf16 sP[4][16][32];     // 4 KiB

    const int tid  = threadIdx.x;
    const int wave = tid >> 5;
    const int lane = tid & 31;
    const int half = lane >> 4;
    const int l16  = lane & 15;
    const int bh   = blockIdx.y;
    const int q0   = blockIdx.x * 64 + wave * 16;
    const float scale = 0.125f;                 // 1/sqrt(64)

    const __bf16* Qh  = Q  + (size_t)bh * NSEQ * HDIM;
    const __bf16* Kh  = K  + (size_t)bh * NSEQ * HDIM;
    const __bf16* Vth = Vt + (size_t)bh * HDIM * NSEQ;

    v16bf qfr[2];
#pragma unroll
    for (int s = 0; s < 2; ++s)
        qfr[s] = *(const v16bf*)&Qh[(size_t)(q0 + l16) * HDIM + s * 32 +
                                    half * 16];

    float mrow[8], lrow[8];
#pragma unroll
    for (int r = 0; r < 8; ++r) { mrow[r] = -1e30f; lrow[r] = 0.0f; }
    v8f oacc[4];
#pragma unroll
    for (int j = 0; j < 4; ++j) oacc[j] = v8f_zero();

    // Stage K (32x8 chunks) and V^T (64x4 chunks): 2 chunks each per thread.
    auto stage = [&](int buf, int kb) {
#pragma unroll
        for (int c = tid; c < 256; c += 128) {
            const int r = c >> 3, cg = (c & 7) * 8;
            async_cp16(Kh + (size_t)(kb + r) * HDIM + cg, &sK[buf][r][cg]);
        }
#pragma unroll
        for (int c = tid; c < 256; c += 128) {
            const int hd = c >> 2, cg = (c & 3) * 8;
            async_cp16(Vth + (size_t)hd * NSEQ + kb + cg, &sVt[buf][hd][cg]);
        }
    };

    stage(0, 0);
    async_wait0();
    __syncthreads();

    int cur = 0;
    for (int kb = 0; kb < NSEQ; kb += 32) {
        if (kb + 32 < NSEQ) stage(cur ^ 1, kb + 32);

        // --- S = scale * Q.K^T (2 column tiles of 16 k-positions) ---
        v8f sacc[2];
#pragma unroll
        for (int j = 0; j < 2; ++j) {
            v16bf b0 = *(const v16bf*)&sK[cur][j * 16 + l16][half * 16];
            v16bf b1 = *(const v16bf*)&sK[cur][j * 16 + l16][32 + half * 16];
            v8f t = v8f_zero();
            t = __builtin_amdgcn_wmma_f32_16x16x32_bf16(false, qfr[0], false,
                                                        b0, (short)0, t,
                                                        false, false);
            t = __builtin_amdgcn_wmma_f32_16x16x32_bf16(false, qfr[1], false,
                                                        b1, (short)0, t,
                                                        false, false);
            sacc[j] = t;
        }

        // --- online softmax (rows 0-7 in lanes 0-15, 8-15 in 16-31) ---
#pragma unroll
        for (int r = 0; r < 8; ++r) {
            float a = sacc[0][r] * scale;
            float b = sacc[1][r] * scale;
            float mx = fmaxf(a, b);
#pragma unroll
            for (int off = 1; off < 16; off <<= 1)
                mx = fmaxf(mx, __shfl_xor(mx, off, 32));
            const float mnew = fmaxf(mrow[r], mx);
            const float corr = __expf(mrow[r] - mnew);
            const float pa = __expf(a - mnew);
            const float pb = __expf(b - mnew);
            float sum = pa + pb;
#pragma unroll
            for (int off = 1; off < 16; off <<= 1)
                sum += __shfl_xor(sum, off, 32);
            lrow[r] = lrow[r] * corr + sum;
            mrow[r] = mnew;
#pragma unroll
            for (int j = 0; j < 4; ++j) oacc[j][r] *= corr;

            const int prow = r + half * 8;
            sP[wave][prow][l16]      = (__bf16)pa;
            sP[wave][prow][16 + l16] = (__bf16)pb;
        }

        // --- O += P.V (single WMMA k-step over the 32 k-positions) ---
        v16bf pfr = *(const v16bf*)&sP[wave][l16][half * 16];
#pragma unroll
        for (int j = 0; j < 4; ++j) {
            v16bf vfr = *(const v16bf*)&sVt[cur][j * 16 + l16][half * 16];
            oacc[j] = __builtin_amdgcn_wmma_f32_16x16x32_bf16(
                false, pfr, false, vfr, (short)0, oacc[j], false, false);
        }

        async_wait0();
        __syncthreads();
        cur ^= 1;
    }

    // --- normalize and store bf16 into (B,N,D) layout ---
    const int h = bh % NHEAD;
    const int b = bh / NHEAD;
#pragma unroll
    for (int j = 0; j < 4; ++j) {
#pragma unroll
        for (int r = 0; r < 8; ++r) {
            const int n = q0 + r + half * 8;
            const int d = h * HDIM + j * 16 + l16;
            O[((size_t)b * NSEQ + n) * DMODEL + d] =
                (__bf16)(oacc[j][r] / lrow[r]);
        }
    }
}

// ---------------------------------------------------------------------------
// Host-side launch
// ---------------------------------------------------------------------------
extern "C" void kernel_launch(void* const* d_in, const int* in_sizes, int n_in,
                              void* d_out, int out_size, void* d_ws,
                              size_t ws_size, hipStream_t stream) {
    (void)in_sizes; (void)n_in; (void)out_size; (void)ws_size;

    const float* x  = (const float*)d_in[0];
    const float* Wq = (const float*)d_in[1];
    const float* bq = (const float*)d_in[2];
    const float* Wk = (const float*)d_in[3];
    const float* bk = (const float*)d_in[4];
    const float* Wv = (const float*)d_in[5];
    const float* bv = (const float*)d_in[6];
    const float* Wo = (const float*)d_in[7];
    const float* bo = (const float*)d_in[8];

    const int BND = BDIM * NSEQ * DMODEL;   // 8,388,608
    const int Mrows = BDIM * NSEQ;          // 8192

    char* ws = (char*)d_ws;
    const size_t SZ_X = (size_t)BND * 2;                 // 16 MiB
    const size_t SZ_W = (size_t)DMODEL * DMODEL * 2;     // 2 MiB
    __bf16* Xbf  = (__bf16*)(ws);
    __bf16* WqT  = (__bf16*)(ws + SZ_X);                 // transposed [N][K]
    __bf16* WkT  = (__bf16*)(ws + SZ_X + SZ_W);
    __bf16* WvT  = (__bf16*)(ws + SZ_X + 2 * SZ_W);
    __bf16* WoT  = (__bf16*)(ws + SZ_X + 3 * SZ_W);
    char*   p    = ws + SZ_X + 4 * SZ_W;
    __bf16* Qh = (__bf16*)(p);              p += SZ_X;   // Q proj (B,N,D)
    __bf16* Kh = (__bf16*)(p);              p += SZ_X;
    __bf16* Vh = (__bf16*)(p);              p += SZ_X;
    __bf16* Qr = (__bf16*)(p);              p += SZ_X;   // roped (B,H,N,HD)
    __bf16* Kr = (__bf16*)(p);              p += SZ_X;
    __bf16* Vt = (__bf16*)(p);              p += SZ_X;   // (B,H,HD,N)
    __bf16* Obf = Xbf;  // Xbf dead after QKV GEMMs -> reuse for O

    // 1) casts + weight transposes
    k_to_bf16<<<(BND + 255) / 256, 256, 0, stream>>>(x, Xbf, BND);
    dim3 gt(DMODEL / 32, DMODEL / 32), bt(32, 8);
    k_transpose_bf16<<<gt, bt, 0, stream>>>(Wq, WqT);
    k_transpose_bf16<<<gt, bt, 0, stream>>>(Wk, WkT);
    k_transpose_bf16<<<gt, bt, 0, stream>>>(Wv, WvT);
    k_transpose_bf16<<<gt, bt, 0, stream>>>(Wo, WoT);

    // 2) QKV projections (bf16 out)
    dim3 gg(DMODEL / GBN, Mrows / GBM);     // (8, 64)
    k_gemm_bf16_wmma<<<gg, 256, 0, stream>>>(Xbf, WqT, bq, Qh,
                                             Mrows, DMODEL, DMODEL, 1);
    k_gemm_bf16_wmma<<<gg, 256, 0, stream>>>(Xbf, WkT, bk, Kh,
                                             Mrows, DMODEL, DMODEL, 1);
    k_gemm_bf16_wmma<<<gg, 256, 0, stream>>>(Xbf, WvT, bv, Vh,
                                             Mrows, DMODEL, DMODEL, 1);

    // 3) RoPE + permutes
    k_rope_permute<<<BDIM * NHEAD * NSEQ, 64, 0, stream>>>(Qh, Kh, Qr, Kr);
    k_permute_vt<<<(BND + 255) / 256, 256, 0, stream>>>(Vh, Vt, BND);

    // 4) attention
    dim3 ga(NSEQ / 64, BDIM * NHEAD);       // (32, 64)
    k_flash_attn_wmma<<<ga, 128, 0, stream>>>(Qr, Kr, Vt, Obf);

    // 5) output projection (fp32 out into d_out)
    k_gemm_bf16_wmma<<<gg, 256, 0, stream>>>(Obf, WoT, bo, (float*)d_out,
                                             Mrows, DMODEL, DMODEL, 0);
}
